// CrossAttention_86019605004951
// MI455X (gfx1250) — matrix operand order repaired and checked by reading, not compiled, not run on hardware
//
#include <hip/hip_runtime.h>
#include <hip/hip_bf16.h>
#include <stdint.h>

typedef __attribute__((ext_vector_type(16))) __bf16 v16bf;
typedef __attribute__((ext_vector_type(8)))  float  v8f;
typedef uint32_t su4 __attribute__((ext_vector_type(4)));
typedef uint32_t su8 __attribute__((ext_vector_type(8)));

#define B_   4
#define T_   2048
#define HIN_ 1024
#define H_   16
#define E_   64
#define HE_  (H_ * E_)   // 1024
#define M_   (B_ * T_)   // 8192

// Native f32->bf16 (lowers to v_fma_mix*_bf16 / packed cvt on gfx1250, RNE)
__device__ __forceinline__ unsigned short bf16_bits(float x) {
    union { __bf16 b; unsigned short u; } c; c.b = (__bf16)x; return c.u;
}
__device__ __forceinline__ uint32_t pack2_bf16(float x, float y) {
    return (uint32_t)bf16_bits(x) | ((uint32_t)bf16_bits(y) << 16);
}

union Frag16 { v16bf v; uint32_t ui[8]; unsigned short us[16]; };
union Acc8   { v8f v; float f[8]; };

// wave32 xor-shuffle via ds_bpermute (no wave64 assumptions)
__device__ __forceinline__ float lane_xor_f(float x, int mask) {
    union { float f; int i; } c; c.f = x;
    c.i = __builtin_amdgcn_ds_bpermute((((int)(threadIdx.x & 31)) ^ mask) << 2, c.i);
    return c.f;
}

// CDNA5 async copy: global -> LDS without VGPR round trip (ASYNCcnt-tracked)
__device__ __forceinline__ void async_load_b128(uint32_t lds_off, uint64_t gaddr) {
    asm volatile("global_load_async_to_lds_b128 %0, %1, off"
                 :: "v"(lds_off), "v"(gaddr) : "memory");
}
__device__ __forceinline__ void wait_async0() {
    asm volatile("s_wait_asynccnt 0x0" ::: "memory");
}

// CDNA5 Tensor Data Mover: one-instruction 2D tile DMA, global -> LDS.
// Tile = `rows` x 128 bytes, row stride 2048 bytes (bf16 head slice of [t][HE]).
// D# per ISA ch.8: data_size=8B units; tensor_dim0=tile_dim0=16 units;
// tensor_dim0_stride=256 units; workgroup_mask=0 (not in a cluster).
__device__ __forceinline__ void tdm_load_tile(uint32_t lds_off, const void* gptr,
                                              uint32_t rows) {
    const uint64_t ga = (uint64_t)(uintptr_t)gptr;
    su4 g0;
    g0[0] = 1u;                                              // count=1
    g0[1] = lds_off;                                         // lds_addr
    g0[2] = (uint32_t)ga;                                    // global_addr[31:0]
    g0[3] = (uint32_t)((ga >> 32) & 0x1FFFFFFu) | (2u << 30);// addr[56:32]|type=2
    su8 g1;
    g1[0] = 3u << 16;                 // data_size = 8 bytes
    g1[1] = 16u << 16;                // tensor_dim0 = 16 units (128B row)
    g1[2] = 0u;                       // tensor_dim0 hi / tensor_dim1 lo
    g1[3] = 0x10u | (16u << 16);      // tensor_dim1 = 1<<20 rows; tile_dim0 = 16
    g1[4] = rows;                     // tile_dim1 = rows, tile_dim2 = 0
    g1[5] = 256u;                     // tensor_dim0_stride = 256 units (2048B)
    g1[6] = 0u;
    g1[7] = 0u;
    asm volatile("tensor_load_to_lds %0, %1" :: "s"(g0), "s"(g1) : "memory");
}

// ---------------------------------------------------------------------------
// Kernel 1: C[M,HE] = X[M,HIN] * W[HIN,HE], output bf16 (scale folded in).
// Block = 256 thr (8 waves), tile 128(M) x 64(N); wave computes 16x64.
// (f32 inputs need conversion en route, so staging stays on the VALU path.)
// ---------------------------------------------------------------------------
__global__ __launch_bounds__(256)
void proj_kernel(const float* __restrict__ X, const float* __restrict__ W,
                 unsigned short* __restrict__ Out, float scale) {
    __shared__ unsigned short ldsA[128][32];  // [m][k] bf16
    __shared__ unsigned short ldsB[64][32];   // [n][k] bf16 (transposed)

    const int tid  = threadIdx.x;
    const int lane = tid & 31;
    const int wave = tid >> 5;
    const int half = lane >> 4;
    const int l16  = lane & 15;
    const int row0 = blockIdx.x * 128;
    const int n0   = blockIdx.y * 64;

    Acc8 acc[4];
    #pragma unroll
    for (int j = 0; j < 4; ++j)
        #pragma unroll
        for (int i = 0; i < 8; ++i) acc[j].f[i] = 0.0f;

    for (int k0 = 0; k0 < HIN_; k0 += 32) {
        __syncthreads();
        #pragma unroll
        for (int it = 0; it < 4; ++it) {
            int g = tid + it * 256;
            int r = (g * 4) >> 5;
            int c = (g * 4) & 31;
            const float4 f4 = *(const float4*)&X[(size_t)(row0 + r) * HIN_ + k0 + c];
            *(uint32_t*)&ldsA[r][c]     = pack2_bf16(f4.x, f4.y);
            *(uint32_t*)&ldsA[r][c + 2] = pack2_bf16(f4.z, f4.w);
        }
        #pragma unroll
        for (int it = 0; it < 2; ++it) {
            int g = tid + it * 256;
            int r = (g * 4) >> 6;
            int c = (g * 4) & 63;
            const float4 f4 = *(const float4*)&W[(size_t)(k0 + r) * HE_ + n0 + c];
            ldsB[c + 0][r] = bf16_bits(f4.x);
            ldsB[c + 1][r] = bf16_bits(f4.y);
            ldsB[c + 2][r] = bf16_bits(f4.z);
            ldsB[c + 3][r] = bf16_bits(f4.w);
        }
        __syncthreads();

        Frag16 a;
        const int mrow = wave * 16 + l16;
        #pragma unroll
        for (int r = 0; r < 8; ++r) {
            int k = ((r >= 4) ? 16 : 0) + half * 8 + (r & 3) * 2;
            a.ui[r] = *(const uint32_t*)&ldsA[mrow][k];
        }
        #pragma unroll
        for (int j = 0; j < 4; ++j) {
            Frag16 b;
            const int nn = j * 16 + l16;
            #pragma unroll
            for (int r = 0; r < 8; ++r) {
                int k = ((r >= 4) ? 16 : 0) + half * 8 + (r & 3) * 2;
                b.ui[r] = *(const uint32_t*)&ldsB[nn][k];
            }
            acc[j].v = __builtin_amdgcn_wmma_f32_16x16x32_bf16(
                false, a.v, false, b.v, (short)0, acc[j].v, false, false);
        }
    }

    #pragma unroll
    for (int j = 0; j < 4; ++j)
        #pragma unroll
        for (int i = 0; i < 8; ++i) {
            int row = row0 + wave * 16 + i + half * 8;
            int col = n0 + j * 16 + l16;
            Out[(size_t)row * HE_ + col] = bf16_bits(acc[j].f[i] * scale);
        }
}

// ---------------------------------------------------------------------------
// Kernel 2: flash attention. Block = 8 waves; wave owns 16 query rows of one
// (b,h). Q block + K tiles staged by the Tensor Data Mover (TENSORcnt);
// V rows streamed with async global->LDS (ASYNCcnt) then LDS-transposed.
// ---------------------------------------------------------------------------
__global__ __launch_bounds__(256)
void attn_kernel(const unsigned short* __restrict__ Qb,
                 const unsigned short* __restrict__ Kb,
                 const unsigned short* __restrict__ Vb,
                 float* __restrict__ Out) {
    __shared__ unsigned short ldsK[64][64];     // [key][e]   (TDM dest)
    __shared__ unsigned short ldsVr[64][64];    // [key][e]   (async dest)
    __shared__ unsigned short ldsV[64][64];     // [e][key]   (transposed)
    __shared__ unsigned short ldsPQ[8][16][64]; // P staging; aliases Q block

    const int tid  = threadIdx.x;
    const int lane = tid & 31;
    const int wave = tid >> 5;
    const int half = lane >> 4;
    const int l16  = lane & 15;
    const int b    = blockIdx.x >> 4;
    const int h    = blockIdx.x & 15;
    const int t0   = blockIdx.y * 128 + wave * 16;
    const size_t headoff = (size_t)h * E_;

    // ---- Q block (128 x 64 bf16) via TDM, then into registers -------------
    unsigned short (*ldsQ)[64] = (unsigned short (*)[64])&ldsPQ[0][0][0];
    if (wave == 0) {
        tdm_load_tile((uint32_t)(uintptr_t)&ldsQ[0][0],
                      &Qb[(size_t)(b * T_ + blockIdx.y * 128) * HE_ + headoff], 128u);
        __builtin_amdgcn_s_wait_tensorcnt(0);
    }
    __syncthreads();
    Frag16 qf[2];
    #pragma unroll
    for (int f = 0; f < 2; ++f)
        #pragma unroll
        for (int r = 0; r < 8; ++r) {
            int k = f * 32 + ((r >= 4) ? 16 : 0) + half * 8 + (r & 3) * 2;
            qf[f].ui[r] = *(const uint32_t*)&ldsQ[wave * 16 + l16][k];
        }
    __syncthreads();   // ldsPQ now free for P staging

    Acc8 O[4];
    #pragma unroll
    for (int j = 0; j < 4; ++j)
        #pragma unroll
        for (int i = 0; i < 8; ++i) O[j].f[i] = 0.0f;
    float m_run[8], l_run[8];
    #pragma unroll
    for (int i = 0; i < 8; ++i) { m_run[i] = -1e30f; l_run[i] = 0.0f; }

    for (int kt = 0; kt < T_; kt += 64) {
        __syncthreads();   // guard LDS reuse across iterations
        // K tile: single TDM DMA from wave 0.
        if (wave == 0)
            tdm_load_tile((uint32_t)(uintptr_t)&ldsK[0][0],
                          &Kb[(size_t)(b * T_ + kt) * HE_ + headoff], 64u);
        // V rows: async copies straight into LDS (row-major scratch).
        #pragma unroll
        for (int it = 0; it < 2; ++it) {
            int g   = tid + it * 256;           // 512 groups of 8 bf16
            int row = g >> 3;
            int e0  = (g & 7) * 8;
            async_load_b128((uint32_t)(uintptr_t)&ldsVr[row][e0],
                            (uint64_t)(uintptr_t)&Vb[(size_t)(b * T_ + kt + row) * HE_ + headoff + e0]);
        }
        if (kt + 64 < T_) {
            __builtin_prefetch(&Kb[(size_t)(b * T_ + kt + 64 + (tid >> 2)) * HE_ + headoff], 0, 0);
            __builtin_prefetch(&Vb[(size_t)(b * T_ + kt + 64 + (tid >> 2)) * HE_ + headoff], 0, 0);
        }
        wait_async0();
        if (wave == 0) __builtin_amdgcn_s_wait_tensorcnt(0);
        __syncthreads();   // ldsK + ldsVr valid

        // LDS->LDS transpose of V into [e][key]
        #pragma unroll
        for (int it = 0; it < 2; ++it) {
            int g   = tid + it * 256;
            int row = g >> 3;
            int e0  = (g & 7) * 8;
            union { uint4 q; unsigned short us[8]; } vv;
            vv.q = *(const uint4*)&ldsVr[row][e0];
            #pragma unroll
            for (int t = 0; t < 8; ++t) ldsV[e0 + t][row] = vv.us[t];
        }
        __syncthreads();   // ldsV valid

        // S(16x64) = q * K^T : 4 N-subtiles x 2 K-halves = 8 WMMAs
        Acc8 S[4];
        #pragma unroll
        for (int j = 0; j < 4; ++j) {
            #pragma unroll
            for (int i = 0; i < 8; ++i) S[j].f[i] = 0.0f;
            const int key = j * 16 + l16;
            #pragma unroll
            for (int f = 0; f < 2; ++f) {
                Frag16 bk;
                #pragma unroll
                for (int r = 0; r < 8; ++r) {
                    int e = f * 32 + ((r >= 4) ? 16 : 0) + half * 8 + (r & 3) * 2;
                    bk.ui[r] = *(const uint32_t*)&ldsK[key][e];
                }
                S[j].v = __builtin_amdgcn_wmma_f32_16x16x32_bf16(
                    false, qf[f].v, false, bk.v, (short)0, S[j].v, false, false);
            }
        }

        // online softmax: acc elem i <-> row (i + 8*half); 16-lane reductions
        #pragma unroll
        for (int i = 0; i < 8; ++i) {
            float mx = fmaxf(fmaxf(S[0].f[i], S[1].f[i]), fmaxf(S[2].f[i], S[3].f[i]));
            #pragma unroll
            for (int d = 1; d < 16; d <<= 1) mx = fmaxf(mx, lane_xor_f(mx, d));
            float m_new = fmaxf(m_run[i], mx);
            float alpha = __expf(m_run[i] - m_new);
            float s = 0.0f;
            #pragma unroll
            for (int j = 0; j < 4; ++j) {
                float p = __expf(S[j].f[i] - m_new);
                S[j].f[i] = p;
                s += p;
            }
            #pragma unroll
            for (int d = 1; d < 16; d <<= 1) s += lane_xor_f(s, d);
            l_run[i] = l_run[i] * alpha + s;
            m_run[i] = m_new;
            #pragma unroll
            for (int j = 0; j < 4; ++j) O[j].f[i] *= alpha;
            const int prow = i + half * 8;
            #pragma unroll
            for (int j = 0; j < 4; ++j)
                ldsPQ[wave][prow][j * 16 + l16] = bf16_bits(S[j].f[i]);
        }

        // O(16x64) += P(16x64) * V(64x64): 2 K-halves x 4 E-subtiles = 8 WMMAs
        Frag16 pf[2];
        #pragma unroll
        for (int f = 0; f < 2; ++f)
            #pragma unroll
            for (int r = 0; r < 8; ++r) {
                int k = f * 32 + ((r >= 4) ? 16 : 0) + half * 8 + (r & 3) * 2;
                pf[f].ui[r] = *(const uint32_t*)&ldsPQ[wave][l16][k];
            }
        #pragma unroll
        for (int j = 0; j < 4; ++j) {
            const int e = j * 16 + l16;
            #pragma unroll
            for (int f = 0; f < 2; ++f) {
                Frag16 bv;
                #pragma unroll
                for (int r = 0; r < 8; ++r) {
                    int key = f * 32 + ((r >= 4) ? 16 : 0) + half * 8 + (r & 3) * 2;
                    bv.ui[r] = *(const uint32_t*)&ldsV[e][key];
                }
                O[j].v = __builtin_amdgcn_wmma_f32_16x16x32_bf16(
                    false, pf[f].v, false, bv.v, (short)0, O[j].v, false, false);
            }
        }
    }

    // normalize + store f32
    #pragma unroll
    for (int i = 0; i < 8; ++i) {
        float inv = 1.0f / l_run[i];
        int trow = t0 + i + half * 8;
        #pragma unroll
        for (int j = 0; j < 4; ++j)
            Out[(size_t)(b * T_ + trow) * HE_ + headoff + j * 16 + l16] = O[j].f[i] * inv;
    }
}

// ---------------------------------------------------------------------------
extern "C" void kernel_launch(void* const* d_in, const int* in_sizes, int n_in,
                              void* d_out, int out_size, void* d_ws, size_t ws_size,
                              hipStream_t stream) {
    const float* query = (const float*)d_in[0];
    const float* keyv  = (const float*)d_in[1];
    const float* Wq    = (const float*)d_in[2];
    const float* Wk    = (const float*)d_in[3];
    const float* Wv    = (const float*)d_in[4];
    float* out = (float*)d_out;

    unsigned short* qb = (unsigned short*)d_ws;           // 3 x 16 MB bf16
    unsigned short* kb = qb + (size_t)M_ * HE_;
    unsigned short* vb = kb + (size_t)M_ * HE_;

    const float scale = 0.3535533905932738f;  // 64^(-0.25), folded into Q and K

    dim3 pgrid(M_ / 128, HE_ / 64);
    proj_kernel<<<pgrid, 256, 0, stream>>>(query, Wq, qb, scale);
    proj_kernel<<<pgrid, 256, 0, stream>>>(keyv, Wk, kb, scale);
    proj_kernel<<<pgrid, 256, 0, stream>>>(keyv, Wv, vb, 1.0f);

    dim3 agrid(B_ * H_, T_ / 128);
    attn_kernel<<<agrid, 256, 0, stream>>>(qb, kb, vb, out);
}
